// BanzhafGuidance_44890998178037
// MI455X (gfx1250) — compile-verified, matrix operand order checked by reading, not run on hardware
//
#include <hip/hip_runtime.h>
#include <hip/hip_bf16.h>
#include <math.h>

typedef __attribute__((ext_vector_type(16))) _Float16 v16h;
typedef __attribute__((ext_vector_type(8)))  _Float16 v8h;
typedef __attribute__((ext_vector_type(8)))  float    v8f;

#define NB   4096      // batch B
#define KD   256       // feature dim D
#define BMT  128       // block tile M
#define BNT  128       // block tile N
#define BKT  32        // k-step (one f16 WMMA)
#define LDST 40        // LDS row stride in f16 (pad 32 -> 40, conflict-free b128 reads)

__device__ __forceinline__ v8f wmma_f16(v16h a, v16h b, v8f c) {
  // D = A*B + C, f16 inputs, f32 accum
  return __builtin_amdgcn_wmma_f32_16x16x32_f16(false, a, false, b, (short)0, c, false, false);
}

// CDNA5 async global->LDS DMA (ASYNCcnt-tracked), bypasses VGPRs.
__device__ __forceinline__ void async_ld16(uint32_t lds_off, const void* gaddr) {
  asm volatile("global_load_async_to_lds_b128 %0, %1, off"
               :: "v"(lds_off), "v"((unsigned long long)(uintptr_t)gaddr)
               : "memory");
}
__device__ __forceinline__ void wait_async0() {
  asm volatile("s_wait_asynccnt 0x0" ::: "memory");
}

// ---------------- K1: L2-normalize rows, split fp32 -> f16 hi/lo -------------
__global__ __launch_bounds__(256) void k_norm_split(
    const float* __restrict__ gI, const float* __restrict__ gT,
    _Float16* __restrict__ Ahi, _Float16* __restrict__ Alo,
    _Float16* __restrict__ Bhi, _Float16* __restrict__ Blo) {
  int row  = blockIdx.x & (NB - 1);
  bool isB = blockIdx.x >= NB;
  const float* src = isB ? gT : gI;
  _Float16* hi = isB ? Bhi : Ahi;
  _Float16* lo = isB ? Blo : Alo;
  int t = threadIdx.x;                       // 256 threads == KD
  float v = src[(size_t)row * KD + t];
  __shared__ float red[256];
  red[t] = v * v;
  __syncthreads();
  for (int s = 128; s > 0; s >>= 1) { if (t < s) red[t] += red[t + s]; __syncthreads(); }
  float inv = 1.0f / fmaxf(sqrtf(red[0]), 1e-12f);
  float x = v * inv;
  _Float16 h = (_Float16)x;
  _Float16 l = (_Float16)(x - (float)h);
  hi[(size_t)row * KD + t] = h;
  lo[(size_t)row * KD + t] = l;
}

// ---------------- K2: S = A * B^T via split-precision f16 WMMA ---------------
__global__ __launch_bounds__(256) void k_gemm_wmma(
    const _Float16* __restrict__ Ahi, const _Float16* __restrict__ Alo,
    const _Float16* __restrict__ Bhi, const _Float16* __restrict__ Blo,
    float* __restrict__ S) {
  // Double-buffered tiles: 2 x 4 x (128*40) f16 = 80 KB LDS
  __shared__ _Float16 sAhi[2][BMT * LDST];
  __shared__ _Float16 sAlo[2][BMT * LDST];
  __shared__ _Float16 sBhi[2][BNT * LDST];
  __shared__ _Float16 sBlo[2][BNT * LDST];

  const int t     = threadIdx.x;
  const int wave  = t >> 5;
  const int lane  = t & 31;
  const int half  = lane >> 4;       // 0: lanes 0-15, 1: lanes 16-31
  const int r     = lane & 15;
  const int waveM = wave >> 2;       // 0..1  (64 rows each)
  const int waveN = wave & 3;        // 0..3  (32 cols each)
  const int blockM = blockIdx.y * BMT;
  const int blockN = blockIdx.x * BNT;

  // Per-thread staging geometry: 512 16B chunks per array, 2 per thread.
  int srow[2], scc[2];
#pragma unroll
  for (int j = 0; j < 2; ++j) { int idx = t + j * 256; srow[j] = idx >> 2; scc[j] = idx & 3; }

  auto stage = [&](int buf, int k0) {
#pragma unroll
    for (int j = 0; j < 2; ++j) {
      int row = srow[j], cc = scc[j];
      size_t ga = (size_t)(blockM + row) * KD + k0 + cc * 8;
      size_t gb = (size_t)(blockN + row) * KD + k0 + cc * 8;
      int lo = row * LDST + cc * 8;
      async_ld16((uint32_t)(uintptr_t)&sAhi[buf][lo], &Ahi[ga]);
      async_ld16((uint32_t)(uintptr_t)&sAlo[buf][lo], &Alo[ga]);
      async_ld16((uint32_t)(uintptr_t)&sBhi[buf][lo], &Bhi[gb]);
      async_ld16((uint32_t)(uintptr_t)&sBlo[buf][lo], &Blo[gb]);
    }
  };

  v8f c[4][2];
  const v8f vzero = {0.f, 0.f, 0.f, 0.f, 0.f, 0.f, 0.f, 0.f};
#pragma unroll
  for (int m = 0; m < 4; ++m)
#pragma unroll
    for (int n = 0; n < 2; ++n) c[m][n] = vzero;

  // Prologue: prefetch first k-slab
  stage(0, 0);
  wait_async0();
  __syncthreads();

  int cur = 0;
  for (int ks = 0; ks < KD / BKT; ++ks) {
    // Prefetch next slab into the other buffer while this one is consumed.
    if (ks + 1 < KD / BKT) stage(cur ^ 1, (ks + 1) * BKT);

    // Fragment K offsets per CDNA5 16-bit A layout:
    // lanes 0-15: K {0..7,16..23}; lanes 16-31: K {8..15,24..31}
    const int kA = 8 * half;
    const int kB = 8 * half + 16;

    const _Float16* bufAhi = sAhi[cur];
    const _Float16* bufAlo = sAlo[cur];
    const _Float16* bufBhi = sBhi[cur];
    const _Float16* bufBlo = sBlo[cur];

    v16h aHi[4], aLo[4], bHi[2], bLo[2];
#pragma unroll
    for (int m = 0; m < 4; ++m) {
      int row = waveM * 64 + m * 16 + r;
      v8h p0 = *(const v8h*)&bufAhi[row * LDST + kA];
      v8h p1 = *(const v8h*)&bufAhi[row * LDST + kB];
      aHi[m] = __builtin_shufflevector(p0, p1, 0,1,2,3,4,5,6,7,8,9,10,11,12,13,14,15);
      v8h q0 = *(const v8h*)&bufAlo[row * LDST + kA];
      v8h q1 = *(const v8h*)&bufAlo[row * LDST + kB];
      aLo[m] = __builtin_shufflevector(q0, q1, 0,1,2,3,4,5,6,7,8,9,10,11,12,13,14,15);
    }
#pragma unroll
    for (int n = 0; n < 2; ++n) {
      int col = waveN * 32 + n * 16 + r;
      v8h p0 = *(const v8h*)&bufBhi[col * LDST + kA];
      v8h p1 = *(const v8h*)&bufBhi[col * LDST + kB];
      bHi[n] = __builtin_shufflevector(p0, p1, 0,1,2,3,4,5,6,7,8,9,10,11,12,13,14,15);
      v8h q0 = *(const v8h*)&bufBlo[col * LDST + kA];
      v8h q1 = *(const v8h*)&bufBlo[col * LDST + kB];
      bLo[n] = __builtin_shufflevector(q0, q1, 0,1,2,3,4,5,6,7,8,9,10,11,12,13,14,15);
    }

    // Split-precision product: hi*hi + hi*lo + lo*hi  (~fp32 accuracy)
#pragma unroll
    for (int m = 0; m < 4; ++m)
#pragma unroll
      for (int n = 0; n < 2; ++n) {
        c[m][n] = wmma_f16(aHi[m], bHi[n], c[m][n]);
        c[m][n] = wmma_f16(aHi[m], bLo[n], c[m][n]);
        c[m][n] = wmma_f16(aLo[m], bHi[n], c[m][n]);
      }

    // Prefetch complete + all waves done with current buffer.
    wait_async0();
    __syncthreads();
    cur ^= 1;
  }

  // Epilogue: C/D layout — lane: N = r; VGPR v: M = v + 8*half
#pragma unroll
  for (int m = 0; m < 4; ++m)
#pragma unroll
    for (int n = 0; n < 2; ++n) {
      int colg = blockN + waveN * 32 + n * 16 + r;
#pragma unroll
      for (int v = 0; v < 8; ++v) {
        int rowg = blockM + waveM * 64 + m * 16 + 8 * half + v;
        S[(size_t)rowg * NB + colg] = c[m][n][v];
      }
    }
}

// ---------------- K3: per-row top-2 (+argmax) of S ---------------------------
__global__ __launch_bounds__(256) void k_top2(
    const float* __restrict__ S, float* __restrict__ m1,
    float* __restrict__ m2, int* __restrict__ idx1) {
  int row = blockIdx.x, t = threadIdx.x;
  const float4* S4 = (const float4*)(S + (size_t)row * NB);
  float b1 = -3.0e38f, b2 = -3.0e38f; int i1 = 0;
  for (int q = t; q < NB / 4; q += 256) {
    float4 v4 = S4[q];
    int jb = q * 4;
#pragma unroll
    for (int e = 0; e < 4; ++e) {
      float v = ((const float*)&v4)[e];
      if (v > b1)      { b2 = b1; b1 = v; i1 = jb + e; }
      else if (v > b2) { b2 = v; }
    }
  }
  __shared__ float sm1[256], sm2[256];
  __shared__ int   si[256];
  sm1[t] = b1; sm2[t] = b2; si[t] = i1;
  __syncthreads();
  for (int s = 128; s > 0; s >>= 1) {
    if (t < s) {
      float am1 = sm1[t], am2 = sm2[t]; int ai = si[t];
      float bm1 = sm1[t + s], bm2 = sm2[t + s]; int bi = si[t + s];
      if (bm1 > am1 || (bm1 == am1 && bi < ai)) {
        sm1[t] = bm1; si[t] = bi; sm2[t] = fmaxf(am1, bm2);
      } else {
        sm2[t] = fmaxf(bm1, am2);
      }
    }
    __syncthreads();
  }
  if (t == 0) { m1[row] = sm1[0]; m2[row] = sm2[0]; idx1[row] = si[0]; }
}

// ---------------- K4a: zero colAdj ------------------------------------------
__global__ void k_zero(float* __restrict__ colAdj) {
  colAdj[blockIdx.x * 256 + threadIdx.x] = 0.0f;
}

// ---------------- K4b: colAdj[idx1[i]] += m1[i]-m2[i] ------------------------
__global__ void k_accum(const float* __restrict__ m1, const float* __restrict__ m2,
                        const int* __restrict__ idx1, float* __restrict__ colAdj) {
  int i = blockIdx.x * 256 + threadIdx.x;
  atomicAdd(&colAdj[idx1[i]], m1[i] - m2[i]);
}

// ---------------- K5: e[j], softmax weights w, top-3 of e --------------------
__device__ __forceinline__ void t3_insert(float* v, int* ix, float nv, int ni) {
  for (int k = 0; k < 3; ++k) {
    if (nv > v[k] || (nv == v[k] && ni < ix[k])) {
      for (int m = 2; m > k; --m) { v[m] = v[m - 1]; ix[m] = ix[m - 1]; }
      v[k] = nv; ix[k] = ni; return;
    }
  }
}

__global__ __launch_bounds__(1024) void k_stats(
    const float* __restrict__ colAdj, const float* __restrict__ m1,
    const float* __restrict__ m2, const int* __restrict__ idx1,
    float* __restrict__ e, float* __restrict__ w_out,
    float* __restrict__ top3v, int* __restrict__ top3i) {
  const float invB1  = 1.0f / (float)(NB - 1);
  const float invBB1 = 1.0f / ((float)NB * (float)(NB - 1));
  int t = threadIdx.x;                         // 1024 threads, single block
  __shared__ float sp[NB];                     // pos/TAU
  __shared__ float red[1024];
  __shared__ float s3v[1024 * 3];
  __shared__ int   s3i[1024 * 3];

  for (int j = t; j < NB; j += 1024) e[j] = -colAdj[j] * invBB1;
  __syncthreads();

  for (int i = t; i < NB; i += 1024) {
    float d   = (m1[i] - m2[i]) * invB1;
    float val = e[i] + (idx1[i] == i ? d : 0.0f);
    val = fminf(10.0f, fmaxf(-10.0f, val));
    sp[i] = val * 5.0f;                        // /TAU, TAU = 0.2
  }
  __syncthreads();

  // max
  float lm = -3.0e38f;
  for (int i = t; i < NB; i += 1024) lm = fmaxf(lm, sp[i]);
  red[t] = lm; __syncthreads();
  for (int s = 512; s > 0; s >>= 1) { if (t < s) red[t] = fmaxf(red[t], red[t + s]); __syncthreads(); }
  float mx = red[0];
  __syncthreads();

  // sum exp
  float ls = 0.0f;
  for (int i = t; i < NB; i += 1024) ls += expf(sp[i] - mx);
  red[t] = ls; __syncthreads();
  for (int s = 512; s > 0; s >>= 1) { if (t < s) red[t] += red[t + s]; __syncthreads(); }
  float inv_den = 1.0f / red[0];
  __syncthreads();

  for (int i = t; i < NB; i += 1024) w_out[i] = expf(sp[i] - mx) * inv_den;

  // top-3 of e
  float tv[3] = {-3.0e38f, -3.0e38f, -3.0e38f};
  int   ti[3] = {0x7fffffff, 0x7fffffff, 0x7fffffff};
  for (int j = t; j < NB; j += 1024) t3_insert(tv, ti, e[j], j);
  for (int k = 0; k < 3; ++k) { s3v[t * 3 + k] = tv[k]; s3i[t * 3 + k] = ti[k]; }
  __syncthreads();
  for (int s = 512; s > 0; s >>= 1) {
    if (t < s) {
      float av[3]; int ai[3];
      for (int k = 0; k < 3; ++k) { av[k] = s3v[t * 3 + k]; ai[k] = s3i[t * 3 + k]; }
      for (int k = 0; k < 3; ++k) t3_insert(av, ai, s3v[(t + s) * 3 + k], s3i[(t + s) * 3 + k]);
      for (int k = 0; k < 3; ++k) { s3v[t * 3 + k] = av[k]; s3i[t * 3 + k] = ai[k]; }
    }
    __syncthreads();
  }
  if (t == 0) for (int k = 0; k < 3; ++k) { top3v[k] = s3v[k]; top3i[k] = s3i[k]; }
}

// ---------------- K6: materialize I, per-row hard_j --------------------------
__global__ __launch_bounds__(256) void k_write_I(
    const float* __restrict__ e, const float* __restrict__ m1,
    const float* __restrict__ m2, const int* __restrict__ idx1,
    const float* __restrict__ top3v, const int* __restrict__ top3i,
    float* __restrict__ I_out, float* __restrict__ hj_out) {
  int row = blockIdx.x, t = threadIdx.x;
  int   j0 = idx1[row];
  float d  = (m1[row] - m2[row]) * (1.0f / (float)(NB - 1));
  float4*       dst = (float4*)(I_out + (size_t)row * NB);
  const float4* ev  = (const float4*)e;
  for (int q = t; q < NB / 4; q += 256) {
    float4 vv = ev[q];
    int base = q * 4;
    if (j0 >= base && j0 < base + 4) ((float*)&vv)[j0 - base] += d;
    dst[q] = vv;
  }
  if (t == 0) {
    // argmax over j != row of e[j] + (j==j0)*d   (diag masked to -1e9)
    float cc = (j0 != row) ? (e[j0] + d) : -3.0e38f;
    float cb = -3.0e38f; int bi = 0x7ffffffe;
    for (int k = 0; k < 3; ++k) {
      int ix = top3i[k];
      if (ix != row && ix != j0) { cb = top3v[k]; bi = ix; break; }
    }
    int hj = (cc > cb || (cc == cb && j0 < bi)) ? j0 : bi;
    hj_out[row] = (float)hj;
  }
}

// ---------------------------------------------------------------------------
extern "C" void kernel_launch(void* const* d_in, const int* in_sizes, int n_in,
                              void* d_out, int out_size, void* d_ws, size_t ws_size,
                              hipStream_t stream) {
  (void)in_sizes; (void)n_in; (void)out_size; (void)ws_size;
  const float* gI = (const float*)d_in[0];
  const float* gT = (const float*)d_in[1];

  float* out   = (float*)d_out;
  float* w_out = out;                                   // [4096]
  float* S_out = out + NB;                              // [4096*4096]
  float* I_out = S_out + (size_t)NB * NB;               // [4096*4096]
  float* hj_out= I_out + (size_t)NB * NB;               // [4096]

  // Stage the hi/lo f16 operands inside the I region (unused until K6).
  _Float16* Ahi = (_Float16*)I_out;
  _Float16* Alo = Ahi + (size_t)NB * KD;
  _Float16* Bhi = Alo + (size_t)NB * KD;
  _Float16* Blo = Bhi + (size_t)NB * KD;

  float* ws     = (float*)d_ws;
  float* m1     = ws;                 // 4096
  float* m2     = ws + NB;            // 4096
  int*   idx1   = (int*)(ws + 2 * NB);// 4096
  float* colAdj = ws + 3 * NB;        // 4096
  float* evec   = ws + 4 * NB;        // 4096
  float* top3v  = ws + 5 * NB;        // 3
  int*   top3i  = (int*)(ws + 5 * NB + 4); // 3

  k_norm_split<<<2 * NB, 256, 0, stream>>>(gI, gT, Ahi, Alo, Bhi, Blo);

  dim3 gemm_grid(NB / BNT, NB / BMT);
  k_gemm_wmma<<<gemm_grid, 256, 0, stream>>>(Ahi, Alo, Bhi, Blo, S_out);

  k_top2<<<NB, 256, 0, stream>>>(S_out, m1, m2, idx1);
  k_zero<<<NB / 256, 256, 0, stream>>>(colAdj);
  k_accum<<<NB / 256, 256, 0, stream>>>(m1, m2, idx1, colAdj);
  k_stats<<<1, 1024, 0, stream>>>(colAdj, m1, m2, idx1, evec, w_out, top3v, top3i);
  k_write_I<<<NB, 256, 0, stream>>>(evec, m1, m2, idx1, top3v, top3i, I_out, hj_out);
}
